// MultiSource_Attention_CRF_30064771072193
// MI455X (gfx1250) — matrix operand order repaired and checked by reading, not compiled, not run on hardware
//
#include <hip/hip_runtime.h>
#include <hip/hip_bf16.h>
#include <math.h>

typedef __attribute__((ext_vector_type(16))) _Float16 v16h;
typedef __attribute__((ext_vector_type(8)))  float    v8f;

#define B_  16
#define S_  256
#define W_  16
#define Dw_ 300
#define Dc_ 50
#define F_  128
#define Dt_ 768
#define P_  256
#define H_  128
#define T_  9
#define BL_ 384
#define NT  4096          // B*S tokens

#if defined(__gfx1250__) && __has_builtin(__builtin_amdgcn_global_load_async_to_lds_b128)
#define HAVE_ASYNC_LDS 1
// exact param types for the async builtin: v4i (GCC vector_size) in AS1 / AS3
typedef int async_v4i __attribute__((vector_size(16)));
typedef __attribute__((address_space(1))) async_v4i* g_v4i_ptr;
typedef __attribute__((address_space(3))) async_v4i* l_v4i_ptr;
#endif

// ---------------------------------------------------------------------------
// Prep / gather kernels
// ---------------------------------------------------------------------------

__global__ void gather_w2v_kernel(const float* __restrict__ tab,
                                  const int* __restrict__ words,
                                  _Float16* __restrict__ out)  // [NT,320]
{
    size_t i = (size_t)blockIdx.x * blockDim.x + threadIdx.x;
    if (i >= (size_t)NT * 320) return;
    int t = (int)(i / 320), k = (int)(i % 320);
    float v = (k < Dw_) ? tab[(size_t)words[t] * Dw_ + k] : 0.f;
    out[i] = (_Float16)v;
}

__global__ void gather_tf_kernel(const float* __restrict__ bert,
                                 const int* __restrict__ maps,
                                 _Float16* __restrict__ out)  // [NT,768]
{
    size_t i = (size_t)blockIdx.x * blockDim.x + threadIdx.x;
    if (i >= (size_t)NT * Dt_) return;
    int t = (int)(i / Dt_), k = (int)(i % Dt_);
    int b = t >> 8;
    out[i] = (_Float16)bert[((size_t)b * BL_ + maps[t]) * Dt_ + k];
}

// im2col for char conv: rows = token*16 + w, cols = c*3 + k (150, padded 160)
__global__ void im2col_kernel(const float* __restrict__ ctab,
                              const int* __restrict__ chars,
                              _Float16* __restrict__ out)  // [65536,160]
{
    size_t i = (size_t)blockIdx.x * blockDim.x + threadIdx.x;
    if (i >= (size_t)NT * W_ * 160) return;
    int e   = (int)(i % 160);
    int row = (int)(i / 160);
    int n = row >> 4, w = row & 15;
    float v = 0.f;
    if (e < Dc_ * 3) {
        int c = e / 3, k = e % 3;
        int wp = w + k - 1;
        if (wp >= 0 && wp < W_)
            v = ctab[(size_t)chars[n * W_ + wp] * Dc_ + c];
    }
    out[i] = (_Float16)v;
}

// conv_w [F,Dc,3] -> B matrix [160,128] f16, B[c*3+k, f] = conv_w[f, c, k]
__global__ void prep_convw_kernel(const float* __restrict__ cw,
                                  _Float16* __restrict__ Bm)
{
    int i = blockIdx.x * blockDim.x + threadIdx.x;
    if (i >= 160 * F_) return;
    int r = i / F_, f = i % F_;
    Bm[i] = (_Float16)((r < Dc_ * 3) ? cw[(size_t)f * (Dc_ * 3) + r] : 0.f);
}

// generic f32 -> f16 with row padding (dst rows = total/cols)
__global__ void cast_pad_f16(const float* __restrict__ src,
                             _Float16* __restrict__ dst,
                             int rows_src, int cols, int total)
{
    int i = blockIdx.x * blockDim.x + threadIdx.x;
    if (i >= total) return;
    int r = i / cols;
    dst[i] = (r < rows_src) ? (_Float16)src[i] : (_Float16)0.f;
}

// h2t_w [256,9] -> [256,16] f16 (pad N), bias -> [16] f32
__global__ void prep_h2t_kernel(const float* __restrict__ w,
                                const float* __restrict__ b,
                                _Float16* __restrict__ Bh,
                                float* __restrict__ b16)
{
    int i = blockIdx.x * blockDim.x + threadIdx.x;
    if (i < 256 * 16) {
        int r = i / 16, c = i % 16;
        Bh[i] = (_Float16)((c < T_) ? w[r * T_ + c] : 0.f);
    }
    if (i < 16) b16[i] = (i < T_) ? b[i] : 0.f;
}

// ---------------------------------------------------------------------------
// Generic WMMA GEMM: C[M,N] = act(A[M,K] @ B[K,N] + bias)
// A,B f16 row-major; C f32. grid = (N/16, M/64), block = (32,4).
// act: 0 = none, 1 = relu, 2 = tanh
// ---------------------------------------------------------------------------
__global__ void gemm_wmma_kernel(const _Float16* __restrict__ A, int lda,
                                 const _Float16* __restrict__ Bm, int ldb,
                                 const float* __restrict__ bias,
                                 float* __restrict__ C, int ldc,
                                 int K, int act)
{
    const int lane = threadIdx.x;
    const int m    = lane & 15;
    const int half = lane >> 4;
    const int col0 = blockIdx.x * 16;
    const int row0 = (blockIdx.y * 4 + threadIdx.y) * 16;

    v8f acc = {};
    const _Float16* Arow = A + (size_t)(row0 + m) * lda;
    for (int k0 = 0; k0 < K; k0 += 32) {
        v16h a, b;
#pragma unroll
        for (int g = 0; g < 8; ++g) {
            int k = (g < 4) ? (8 * half + 2 * g) : (16 + 8 * half + 2 * (g - 4));
            a[2 * g]     = Arow[k0 + k];
            a[2 * g + 1] = Arow[k0 + k + 1];
        }
        const _Float16* Brow = Bm + (size_t)k0 * ldb + col0 + m;
#pragma unroll
        for (int e = 0; e < 16; ++e) b[e] = Brow[(size_t)((e < 8) ? (8 * half + e) : (16 + 8 * half + e - 8)) * ldb];
        acc = __builtin_amdgcn_wmma_f32_16x16x32_f16(false, a, false, b,
                                                     (short)0, acc, false, false);
    }
    float bv = bias ? bias[col0 + m] : 0.f;
#pragma unroll
    for (int r = 0; r < 8; ++r) {
        float v = acc[r] + bv;
        if (act == 1)      v = fmaxf(v, 0.f);
        else if (act == 2) v = tanhf(v);
        C[(size_t)(row0 + r + 8 * half) * ldc + col0 + m] = v;
    }
}

// ---------------------------------------------------------------------------
// maxpool over conv width: convout [NT*16,128] -> char_feat f16 [NT,128]
// ---------------------------------------------------------------------------
__global__ void maxpool_kernel(const float* __restrict__ co,
                               _Float16* __restrict__ cf)
{
    int i = blockIdx.x * blockDim.x + threadIdx.x;
    if (i >= NT * F_) return;
    int n = i / F_, f = i % F_;
    float m = co[(size_t)(n * 16) * F_ + f];
    for (int w = 1; w < 16; ++w)
        m = fmaxf(m, co[(size_t)(n * 16 + w) * F_ + f]);
    cf[i] = (_Float16)m;
}

// stacked f16 [NT*3, 256] from p1/p2/p3 f32
__global__ void pack_stacked_kernel(const float* __restrict__ p1,
                                    const float* __restrict__ p2,
                                    const float* __restrict__ p3,
                                    _Float16* __restrict__ st)
{
    size_t i = (size_t)blockIdx.x * blockDim.x + threadIdx.x;
    if (i >= (size_t)NT * 3 * P_) return;
    int col = (int)(i % P_);
    int row = (int)(i / P_);
    int t = row / 3, j = row % 3;
    const float* p = (j == 0) ? p1 : (j == 1) ? p2 : p3;
    st[i] = (_Float16)p[(size_t)t * P_ + col];
}

// per-token attention: scores from projout, softmax over 3 sources, fuse
__global__ void attn_finalize_kernel(const float* __restrict__ proj,  // [NT*3,128]
                                     const float* __restrict__ sw,    // [128]
                                     const float* __restrict__ sb,    // [1]
                                     const float* __restrict__ p1,
                                     const float* __restrict__ p2,
                                     const float* __restrict__ p3,
                                     float* __restrict__ fused,
                                     _Float16* __restrict__ fusedh)
{
    const int t = blockIdx.x;
    const int h = threadIdx.x;  // 0..127
    __shared__ float red[3][128];
    __shared__ float w3[3];
    float swv = sw[h];
#pragma unroll
    for (int j = 0; j < 3; ++j)
        red[j][h] = proj[((size_t)t * 3 + j) * 128 + h] * swv;
    __syncthreads();
    for (int off = 64; off > 0; off >>= 1) {
        if (h < off) {
            red[0][h] += red[0][h + off];
            red[1][h] += red[1][h + off];
            red[2][h] += red[2][h + off];
        }
        __syncthreads();
    }
    if (h == 0) {
        float s0 = red[0][0] + sb[0], s1 = red[1][0] + sb[0], s2 = red[2][0] + sb[0];
        float mx = fmaxf(s0, fmaxf(s1, s2));
        float e0 = expf(s0 - mx), e1 = expf(s1 - mx), e2 = expf(s2 - mx);
        float inv = 1.f / (e0 + e1 + e2);
        w3[0] = e0 * inv; w3[1] = e1 * inv; w3[2] = e2 * inv;
    }
    __syncthreads();
#pragma unroll
    for (int cc = 0; cc < 2; ++cc) {
        int col = h + 128 * cc;
        size_t idx = (size_t)t * P_ + col;
        float v = w3[0] * p1[idx] + w3[1] * p2[idx] + w3[2] * p3[idx];
        fused[idx]  = v;
        fusedh[idx] = (_Float16)v;
    }
}

// ---------------------------------------------------------------------------
// LSTM recurrence with WMMA: one block per direction, 512 threads (16 waves).
// whh (128 KB f16) is staged once into LDS (320 KB/WGP on CDNA5) via the
// gfx1250 async-to-LDS path when available; per step the recurrent matmul
// g[16,512] = pre[:,ts,:] + h[16,128] @ whh[128,512] runs on WMMA with all
// operands in LDS, then gate nonlinearities update h (f16) / c (f32).
// ---------------------------------------------------------------------------
__global__ void lstm_wmma_kernel(const float* __restrict__ pre_f,   // [NT,512]
                                 const float* __restrict__ pre_b,
                                 const _Float16* __restrict__ whh_f, // [128,512]
                                 const _Float16* __restrict__ whh_b,
                                 float* __restrict__ hf,             // [NT,128]
                                 float* __restrict__ hb)
{
    const int dir = blockIdx.x;
    const float*    pre = (dir == 0) ? pre_f : pre_b;
    const _Float16* whh = (dir == 0) ? whh_f : whh_b;
    float*          hs  = (dir == 0) ? hf : hb;

    __shared__ _Float16 whh_s[128 * 512]; // 128 KB
    __shared__ _Float16 h_s[16 * 128];    //   4 KB
    __shared__ float    c_s[16 * 128];    //   8 KB
    __shared__ float    g_s[16 * 512];    //  32 KB   (172 KB total, < 320 KB WGP LDS)

    const int tid  = threadIdx.x;        // 0..511
    const int lane = tid & 31;
    const int wave = tid >> 5;           // 0..15
    const int m    = lane & 15;
    const int half = lane >> 4;

    // ---- stage whh into LDS (4096 x 16B chunks, 8 per thread) ----
    {
        const uint4* src = (const uint4*)whh;
        uint4*       dst = (uint4*)whh_s;
#if defined(HAVE_ASYNC_LDS)
        for (int i = tid; i < 4096; i += 512)
            __builtin_amdgcn_global_load_async_to_lds_b128(
                (g_v4i_ptr)(src + i), (l_v4i_ptr)(dst + i), 0, 0);
#if __has_builtin(__builtin_amdgcn_s_wait_asynccnt)
        __builtin_amdgcn_s_wait_asynccnt(0);
#else
        asm volatile("s_wait_asynccnt 0x0" ::: "memory");
#endif
#else
        for (int i = tid; i < 4096; i += 512) dst[i] = src[i];
#endif
    }
    for (int i = tid; i < 16 * 128; i += 512) {
        h_s[i] = (_Float16)0.f;
        c_s[i] = 0.f;
    }
    __syncthreads();

    for (int s = 0; s < S_; ++s) {
        const int ts = (dir == 0) ? s : (S_ - 1 - s);
        // prefetch next step's pre rows (global_prefetch_b8)
        if (s + 1 < S_) {
            const int tsn = (dir == 0) ? (s + 1) : (S_ - 2 - s);
            __builtin_prefetch(&pre[((size_t)wave * S_ + tsn) * 512 + lane * 16], 0, 0);
        }
#pragma unroll
        for (int j = 0; j < 2; ++j) {
            const int col0 = (wave + 16 * j) * 16;
            v8f acc;
#pragma unroll
            for (int r = 0; r < 8; ++r)
                acc[r] = pre[((size_t)(r + 8 * half) * S_ + ts) * 512 + col0 + m];
#pragma unroll
            for (int kt = 0; kt < 4; ++kt) {
                const int k0 = kt * 32;
                v16h a, b;
#pragma unroll
                for (int g = 0; g < 8; ++g) {
                    int k = (g < 4) ? (8 * half + 2 * g)
                                    : (16 + 8 * half + 2 * (g - 4));
                    a[2 * g]     = h_s[m * 128 + k0 + k];
                    a[2 * g + 1] = h_s[m * 128 + k0 + k + 1];
                }
                const _Float16* Brow = whh_s + (size_t)k0 * 512 + col0 + m;
#pragma unroll
                for (int e = 0; e < 16; ++e) b[e] = Brow[((e < 8) ? (8 * half + e) : (16 + 8 * half + e - 8)) * 512];
                acc = __builtin_amdgcn_wmma_f32_16x16x32_f16(false, a, false, b,
                                                             (short)0, acc,
                                                             false, false);
            }
#pragma unroll
            for (int r = 0; r < 8; ++r)
                g_s[(r + 8 * half) * 512 + col0 + m] = acc[r];
        }
        __syncthreads();
#pragma unroll
        for (int q = 0; q < 4; ++q) {
            const int cell = tid + 512 * q;        // 0..2047
            const int bb = cell >> 7, hh = cell & 127;
            float gi = g_s[bb * 512 + hh];
            float gf = g_s[bb * 512 + 128 + hh];
            float gg = g_s[bb * 512 + 256 + hh];
            float go = g_s[bb * 512 + 384 + hh];
            float c  = c_s[cell];
            float si = 1.f / (1.f + expf(-gi));
            float sf = 1.f / (1.f + expf(-gf));
            float so = 1.f / (1.f + expf(-go));
            c = sf * c + si * tanhf(gg);
            float hv = so * tanhf(c);
            c_s[cell] = c;
            h_s[cell] = (_Float16)hv;
            hs[((size_t)bb * S_ + ts) * 128 + hh] = hv;
        }
        __syncthreads();
    }
}

// concat hf|hb -> f16 [NT,256]
__global__ void concat_kernel(const float* __restrict__ hf,
                              const float* __restrict__ hb,
                              _Float16* __restrict__ out)
{
    size_t i = (size_t)blockIdx.x * blockDim.x + threadIdx.x;
    if (i >= (size_t)NT * 256) return;
    int t = (int)(i >> 8), j = (int)(i & 255);
    float v = (j < 128) ? hf[(size_t)t * 128 + j] : hb[(size_t)t * 128 + j - 128];
    out[i] = (_Float16)v;
}

// ---------------------------------------------------------------------------
// CRF forward algorithm (mask is all-ones). One block per batch, 32 threads.
// em is [B*S,16] with first 9 cols valid.
// ---------------------------------------------------------------------------
__global__ void crf_kernel(const float* __restrict__ em,
                           const int* __restrict__ tags,
                           const float* __restrict__ start,
                           const float* __restrict__ endv,
                           const float* __restrict__ trans,
                           float* __restrict__ llh)
{
    const int b = blockIdx.x;
    const int t = threadIdx.x;
    __shared__ float tr[81];
    __shared__ float alpha[T_];
    __shared__ float score_s;
    __shared__ int   prev_s;
    for (int i = t; i < 81; i += 32) tr[i] = trans[i];
    const float* emb = em + (size_t)b * S_ * 16;
    if (t < T_) alpha[t] = start[t] + emb[t];
    if (t == 0) {
        int tg0 = tags[b * S_];
        score_s = start[tg0] + emb[tg0];
        prev_s  = tg0;
    }
    __syncthreads();
    for (int s = 1; s < S_; ++s) {
        const float* e = emb + s * 16;
        float nv = 0.f;
        if (t < T_) {
            float mx = -3.4e38f;
#pragma unroll
            for (int tt = 0; tt < T_; ++tt)
                mx = fmaxf(mx, alpha[tt] + tr[tt * T_ + t]);
            float sum = 0.f;
#pragma unroll
            for (int tt = 0; tt < T_; ++tt)
                sum += expf(alpha[tt] + tr[tt * T_ + t] - mx);
            nv = mx + logf(sum) + e[t];
        }
        if (t == 0) {
            int tg = tags[b * S_ + s];
            score_s += tr[prev_s * T_ + tg] + e[tg];
            prev_s = tg;
        }
        __syncthreads();
        if (t < T_) alpha[t] = nv;
        __syncthreads();
    }
    if (t == 0) {
        float sc = score_s + endv[prev_s];
        float mx = -3.4e38f;
        for (int tt = 0; tt < T_; ++tt) mx = fmaxf(mx, alpha[tt] + endv[tt]);
        float sum = 0.f;
        for (int tt = 0; tt < T_; ++tt) sum += expf(alpha[tt] + endv[tt] - mx);
        llh[b] = sc - (mx + logf(sum));
    }
}

__global__ void finalize_kernel(const float* __restrict__ llh, float* __restrict__ out)
{
    if (blockIdx.x == 0 && threadIdx.x == 0) {
        float s = 0.f;
        for (int b = 0; b < B_; ++b) s += llh[b];
        out[0] = -s / (float)B_;
    }
}

// ---------------------------------------------------------------------------
extern "C" void kernel_launch(void* const* d_in, const int* in_sizes, int n_in,
                              void* d_out, int out_size, void* d_ws, size_t ws_size,
                              hipStream_t stream)
{
    const int*   words      = (const int*)d_in[0];
    const int*   chars      = (const int*)d_in[1];
    const int*   tags       = (const int*)d_in[2];
    /* d_in[3] word_mask: all-ones, ignored */
    const int*   word_maps  = (const int*)d_in[4];
    const float* bert       = (const float*)d_in[5];
    const float* w2v_table  = (const float*)d_in[6];
    const float* char_table = (const float*)d_in[7];
    const float* conv_w     = (const float*)d_in[8];
    const float* conv_b     = (const float*)d_in[9];
    const float* w2v_pw     = (const float*)d_in[10];
    const float* w2v_pb     = (const float*)d_in[11];
    const float* char_pw    = (const float*)d_in[12];
    const float* char_pb    = (const float*)d_in[13];
    const float* trans_pw   = (const float*)d_in[14];
    const float* trans_pb   = (const float*)d_in[15];
    const float* attn_pw    = (const float*)d_in[16];
    const float* attn_pb    = (const float*)d_in[17];
    const float* attn_sw    = (const float*)d_in[18];
    const float* attn_sb    = (const float*)d_in[19];
    const float* lf_wih     = (const float*)d_in[20];
    const float* lf_whh     = (const float*)d_in[21];
    const float* lf_b       = (const float*)d_in[22];
    const float* lb_wih     = (const float*)d_in[23];
    const float* lb_whh     = (const float*)d_in[24];
    const float* lb_b       = (const float*)d_in[25];
    const float* h2t_w      = (const float*)d_in[26];
    const float* h2t_b      = (const float*)d_in[27];
    const float* crf_start  = (const float*)d_in[28];
    const float* crf_end    = (const float*)d_in[29];
    const float* crf_trans  = (const float*)d_in[30];

    char* ws = (char*)d_ws;
    size_t off = 0;
    auto alloc = [&](size_t bytes) -> void* {
        void* p = ws + off;
        off = (off + bytes + 255) & ~(size_t)255;
        return p;
    };

    _Float16* A1        = (_Float16*)alloc((size_t)NT * 320 * 2);
    _Float16* tfh       = (_Float16*)alloc((size_t)NT * Dt_ * 2);
    _Float16* im2c      = (_Float16*)alloc((size_t)NT * 16 * 160 * 2);
    _Float16* convB     = (_Float16*)alloc(160 * 128 * 2);
    float*    convout   = (float*)alloc((size_t)NT * 16 * 128 * 4);
    _Float16* charf     = (_Float16*)alloc((size_t)NT * 128 * 2);
    _Float16* w2v_pw_h  = (_Float16*)alloc(320 * 256 * 2);
    _Float16* char_pw_h = (_Float16*)alloc(128 * 256 * 2);
    _Float16* trans_pw_h= (_Float16*)alloc(768 * 256 * 2);
    _Float16* attn_pw_h = (_Float16*)alloc(256 * 128 * 2);
    _Float16* lf_wih_h  = (_Float16*)alloc(256 * 512 * 2);
    _Float16* lb_wih_h  = (_Float16*)alloc(256 * 512 * 2);
    _Float16* lf_whh_h  = (_Float16*)alloc(128 * 512 * 2);
    _Float16* lb_whh_h  = (_Float16*)alloc(128 * 512 * 2);
    _Float16* h2t_h     = (_Float16*)alloc(256 * 16 * 2);
    float*    h2t_b16   = (float*)alloc(16 * 4);
    float*    p1        = (float*)alloc((size_t)NT * 256 * 4);
    float*    p2        = (float*)alloc((size_t)NT * 256 * 4);
    float*    p3        = (float*)alloc((size_t)NT * 256 * 4);
    _Float16* stackedh  = (_Float16*)alloc((size_t)NT * 3 * 256 * 2);
    float*    projout   = (float*)alloc((size_t)NT * 3 * 128 * 4);
    float*    fused     = (float*)alloc((size_t)NT * 256 * 4);
    _Float16* fusedh    = (_Float16*)alloc((size_t)NT * 256 * 2);
    float*    pre_f     = (float*)alloc((size_t)NT * 512 * 4);
    float*    pre_b     = (float*)alloc((size_t)NT * 512 * 4);
    float*    hfv       = (float*)alloc((size_t)NT * 128 * 4);
    float*    hbv       = (float*)alloc((size_t)NT * 128 * 4);
    _Float16* lstmh     = (_Float16*)alloc((size_t)NT * 256 * 2);
    float*    emis      = (float*)alloc((size_t)NT * 16 * 4);
    float*    llh       = (float*)alloc(16 * 4);

    auto nb = [](size_t n) { return dim3((unsigned)((n + 255) / 256)); };

    // --- gathers / prep ---
    gather_w2v_kernel<<<nb((size_t)NT * 320), 256, 0, stream>>>(w2v_table, words, A1);
    gather_tf_kernel<<<nb((size_t)NT * Dt_), 256, 0, stream>>>(bert, word_maps, tfh);
    im2col_kernel<<<nb((size_t)NT * 16 * 160), 256, 0, stream>>>(char_table, chars, im2c);
    prep_convw_kernel<<<nb(160 * 128), 256, 0, stream>>>(conv_w, convB);
    cast_pad_f16<<<nb(320 * 256), 256, 0, stream>>>(w2v_pw, w2v_pw_h, 300, 256, 320 * 256);
    cast_pad_f16<<<nb(128 * 256), 256, 0, stream>>>(char_pw, char_pw_h, 128, 256, 128 * 256);
    cast_pad_f16<<<nb(768 * 256), 256, 0, stream>>>(trans_pw, trans_pw_h, 768, 256, 768 * 256);
    cast_pad_f16<<<nb(256 * 128), 256, 0, stream>>>(attn_pw, attn_pw_h, 256, 128, 256 * 128);
    cast_pad_f16<<<nb(256 * 512), 256, 0, stream>>>(lf_wih, lf_wih_h, 256, 512, 256 * 512);
    cast_pad_f16<<<nb(256 * 512), 256, 0, stream>>>(lb_wih, lb_wih_h, 256, 512, 256 * 512);
    cast_pad_f16<<<nb(128 * 512), 256, 0, stream>>>(lf_whh, lf_whh_h, 128, 512, 128 * 512);
    cast_pad_f16<<<nb(128 * 512), 256, 0, stream>>>(lb_whh, lb_whh_h, 128, 512, 128 * 512);
    prep_h2t_kernel<<<nb(256 * 16), 256, 0, stream>>>(h2t_w, h2t_b, h2t_h, h2t_b16);

    dim3 wblk(32, 4);

    // --- char conv as GEMM [65536,160]x[160,128], then maxpool ---
    gemm_wmma_kernel<<<dim3(8, 1024), wblk, 0, stream>>>(im2c, 160, convB, 128,
                                                         conv_b, convout, 128, 160, 0);
    maxpool_kernel<<<nb((size_t)NT * 128), 256, 0, stream>>>(convout, charf);

    // --- three source projections (relu) ---
    gemm_wmma_kernel<<<dim3(16, 64), wblk, 0, stream>>>(A1, 320, w2v_pw_h, 256,
                                                        w2v_pb, p1, 256, 320, 1);
    gemm_wmma_kernel<<<dim3(16, 64), wblk, 0, stream>>>(charf, 128, char_pw_h, 256,
                                                        char_pb, p2, 256, 128, 1);
    gemm_wmma_kernel<<<dim3(16, 64), wblk, 0, stream>>>(tfh, 768, trans_pw_h, 256,
                                                        trans_pb, p3, 256, 768, 1);

    // --- attention: proj = tanh(stacked @ attn_pw + b), then softmax fuse ---
    pack_stacked_kernel<<<nb((size_t)NT * 3 * 256), 256, 0, stream>>>(p1, p2, p3, stackedh);
    gemm_wmma_kernel<<<dim3(8, 192), wblk, 0, stream>>>(stackedh, 256, attn_pw_h, 128,
                                                        attn_pb, projout, 128, 256, 2);
    attn_finalize_kernel<<<NT, 128, 0, stream>>>(projout, attn_sw, attn_sb,
                                                 p1, p2, p3, fused, fusedh);

    // --- LSTM input GEMMs ---
    gemm_wmma_kernel<<<dim3(32, 64), wblk, 0, stream>>>(fusedh, 256, lf_wih_h, 512,
                                                        lf_b, pre_f, 512, 256, 0);
    gemm_wmma_kernel<<<dim3(32, 64), wblk, 0, stream>>>(fusedh, 256, lb_wih_h, 512,
                                                        lb_b, pre_b, 512, 256, 0);

    // --- LSTM recurrence (WMMA in step loop, whh + state LDS-resident) ---
    lstm_wmma_kernel<<<2, 512, 0, stream>>>(pre_f, pre_b, lf_whh_h, lb_whh_h, hfv, hbv);

    // --- emissions GEMM [4096,256]x[256,16] ---
    concat_kernel<<<nb((size_t)NT * 256), 256, 0, stream>>>(hfv, hbv, lstmh);
    gemm_wmma_kernel<<<dim3(1, 64), wblk, 0, stream>>>(lstmh, 256, h2t_h, 16,
                                                       h2t_b16, emis, 16, 256, 0);

    // --- CRF + final reduction ---
    crf_kernel<<<B_, 32, 0, stream>>>(emis, tags, crf_start, crf_end, crf_trans, llh);
    finalize_kernel<<<1, 1, 0, stream>>>(llh, (float*)d_out);
}
